// Attention_3032246911055
// MI455X (gfx1250) — matrix operand order repaired and checked by reading, not compiled, not run on hardware
//
#include <hip/hip_runtime.h>
#include <hip/hip_bf16.h>
#include <math.h>

// ---------------------------------------------------------------------------
// Fused LeViT-attention block for MI455X (gfx1250, wave32, WMMA 16x16x32 f16)
//   B=2048 batch, N=49 tokens (pad 64), DIM=384, H=8 heads, KD=32, VD=512
// One workgroup (256 thr = 8 waves) per batch element, everything staged in
// 192KB static LDS with phase-based region reuse.
// ---------------------------------------------------------------------------

typedef __attribute__((ext_vector_type(16))) _Float16 v16h;
typedef __attribute__((ext_vector_type(8)))  float    v8f;

#define NTOK   49
#define NPAD   64
#define DIMX   384
#define HEADS  8
#define KDIM   32
#define VDIM   512
#define SCALEQ 0.17677669529663687f   // 32^-0.5

// ws layout (bytes)
#define WS_PW    0                         // 1024*384 f16 = 786432
#define WS_PROJ  786432                    // 384*512  f16 = 393216
#define WS_BIAS  (786432 + 393216)         // 8*64*64 f32   = 131072 (layout [h][nk][mq])

struct Frag16 { union { v16h h; unsigned u[8]; }; };

// A-matrix 16x32 f16 fragment (ISA 7.12.2): lane m=l&15, hi=l>>4
//   V0..3 : K = 2j + 8*hi          (K 0..15)
//   V4..7 : K = 16 + 2(j-4) + 8*hi (K 16..31)
__device__ __forceinline__ v16h load_afrag(const _Float16* base, int ldk, int lane) {
    int m = lane & 15, hi = lane >> 4;
    const _Float16* row = base + m * ldk;
    Frag16 f;
#pragma unroll
    for (int j = 0; j < 8; ++j) {
        int k0 = (j < 4) ? (2 * j + 8 * hi) : (16 + 2 * (j - 4) + 8 * hi);
        f.u[j] = *(const unsigned*)(row + k0);
    }
    return f.h;
}

// B-matrix 32x16 f16 fragment: lane n=l&15, hi=l>>4; same K order as A-fragment.
// Source is B^T row-major: [n][K], K contiguous.
__device__ __forceinline__ v16h load_bfrag(const _Float16* base, int ldk, int lane) {
    int n = lane & 15, hi = lane >> 4;
    const _Float16* row = base + n * ldk;
    Frag16 f;
#pragma unroll
    for (int j = 0; j < 8; ++j) {
        int k0 = (j < 4) ? (2 * j + 8 * hi) : (16 + 2 * (j - 4) + 8 * hi);
        f.u[j] = *(const unsigned*)(row + k0);
    }
    return f.h;
}

__device__ __forceinline__ v8f wmma16(v16h a, v16h b, v8f c) {
    return __builtin_amdgcn_wmma_f32_16x16x32_f16(false, a, false, b, (short)0, c, false, false);
}

// ---------------------------------------------------------------------------
// Prep: convert weights to f16 in ws, build padded+masked TRANSPOSED bias
// table: biasfull[h][nk][mq]  (so per-lane loads over mq are contiguous)
// ---------------------------------------------------------------------------
__global__ __launch_bounds__(256) void prep_kernel(
    const float* __restrict__ pw, const float* __restrict__ proj,
    const float* __restrict__ attn_biases, const int* __restrict__ bias_idxs,
    _Float16* __restrict__ pwf16, _Float16* __restrict__ projf16,
    float* __restrict__ biasfull) {
    int i = blockIdx.x * blockDim.x + threadIdx.x;
    if (i < 1024 * DIMX) pwf16[i] = (_Float16)pw[i];
    if (i < DIMX * VDIM) projf16[i] = (_Float16)proj[i];
    if (i < HEADS * NPAD * NPAD) {
        int h = i >> 12, nk = (i >> 6) & 63, mq = i & 63;
        float v;
        if (nk >= NTOK)      v = -1e30f;            // mask padded keys
        else if (mq >= NTOK) v = 0.0f;              // keep padded query rows finite
        else                 v = attn_biases[h * NTOK + bias_idxs[mq * NTOK + nk]];
        biasfull[i] = v;
    }
}

// ---------------------------------------------------------------------------
// Main fused kernel: one block per batch element
// LDS regions (f16 element offsets), 98304 halfs = 192KB total:
//   R0 [0,32768)     : x (A/B phases, [64][384])  ->  v2T (C..E, [8][64][64])
//   R1 [32768,65536) : q [8][64][32] + k [8][64][32] -> attnOut [64][512] (E,F)
//   R2 [65536,98304) : v1 [512][64]               ->  attnP [8][64][64] (D,E)
// ---------------------------------------------------------------------------
__global__ __launch_bounds__(256, 1) void levit_attn_kernel(
    const float* __restrict__ x,
    const float* __restrict__ pwb,
    const float* __restrict__ dww,
    const float* __restrict__ bng, const float* __restrict__ bnb,
    const float* __restrict__ bnm, const float* __restrict__ bnv,
    const _Float16* __restrict__ pwf16,
    const _Float16* __restrict__ projf16,
    const float* __restrict__ biasfull,
    const float* __restrict__ projb,
    float* __restrict__ out) {

    __shared__ _Float16 smem[98304];

    _Float16* xlds    = smem;            // [64][384]  (phase A/B)
    _Float16* v2T     = smem;            // [8][64][64] (phase C..E)
    _Float16* qlds    = smem + 32768;    // [8][64][32]
    _Float16* klds    = smem + 32768 + 16384;
    _Float16* attnOut = smem + 32768;    // [64][512]  (phase E/F)
    _Float16* v1lds   = smem + 65536;    // [512][64]
    _Float16* attnP   = smem + 65536;    // [8][64][64]

    const int tid  = threadIdx.x;
    const int lane = tid & 31;
    const int wave = tid >> 5;
    const int hi   = lane >> 4;
    const int n15  = lane & 15;
    const int blk  = blockIdx.x;

    // ---------------- Phase A: stage x (f32->f16, vectorized), pad ----------
    {
        const float4* xb4 = (const float4*)(x + (size_t)blk * NTOK * DIMX);
        for (int i = tid; i < (NTOK * DIMX) / 4; i += 256) {
            float4 v = xb4[i];
            union { _Float16 h[4]; unsigned long long u; } p;
            p.h[0] = (_Float16)v.x; p.h[1] = (_Float16)v.y;
            p.h[2] = (_Float16)v.z; p.h[3] = (_Float16)v.w;
            *(unsigned long long*)(xlds + i * 4) = p.u;
        }
        for (int i = tid; i < ((NPAD - NTOK) * DIMX) / 4; i += 256)
            *(unsigned long long*)(xlds + NTOK * DIMX + i * 4) = 0ull;
    }
    __syncthreads();

    // ---------------- Phase B: pwconv GEMM  h = W(1024x384) @ x^T -----------
    // 64 M-tiles processed in pairs per wave (B-fragments shared), 4 N-tiles,
    // 12 K-steps. Epilogue branch is tile-uniform (scalar).
    for (int i = 0; i < 4; ++i) {
        int mt0 = wave + 16 * i;
        int mt1 = mt0 + 8;
        // prefetch next pair's weight tiles (global_prefetch_b8)
        if (i < 3) {
            __builtin_prefetch(pwf16 + (mt0 + 16) * 16 * DIMX + lane * 64, 0, 0);
            __builtin_prefetch(pwf16 + (mt1 + 16) * 16 * DIMX + lane * 64, 0, 0);
        }
        v8f acc0[4] = {}, acc1[4] = {};
        for (int kk = 0; kk < 12; ++kk) {
            v16h a0 = load_afrag(pwf16 + mt0 * 16 * DIMX + kk * 32, DIMX, lane);
            v16h a1 = load_afrag(pwf16 + mt1 * 16 * DIMX + kk * 32, DIMX, lane);
#pragma unroll
            for (int nt = 0; nt < 4; ++nt) {
                v16h bb = load_bfrag(xlds + nt * 16 * DIMX + kk * 32, DIMX, lane);
                acc0[nt] = wmma16(a0, bb, acc0[nt]);
                acc1[nt] = wmma16(a1, bb, acc1[nt]);
            }
        }
#pragma unroll
        for (int half = 0; half < 2; ++half) {
            int mt = half ? mt1 : mt0;
            v8f* acc = half ? acc1 : acc0;
            float biasr[8];
#pragma unroll
            for (int r = 0; r < 8; ++r) biasr[r] = pwb[mt * 16 + r + 8 * hi];
            if (mt < 16) {                            // q: o in [0,256)
#pragma unroll
                for (int nt = 0; nt < 4; ++nt) {
                    int n = nt * 16 + n15;
#pragma unroll
                    for (int r = 0; r < 8; ++r) {
                        int o = mt * 16 + r + 8 * hi;
                        qlds[(o >> 5) * 2048 + n * 32 + (o & 31)] =
                            (_Float16)(acc[nt][r] + biasr[r]);
                    }
                }
            } else if (mt < 32) {                     // k: o in [256,512)
#pragma unroll
                for (int nt = 0; nt < 4; ++nt) {
                    int n = nt * 16 + n15;
#pragma unroll
                    for (int r = 0; r < 8; ++r) {
                        int oo = mt * 16 + r + 8 * hi - 256;
                        klds[(oo >> 5) * 2048 + n * 32 + (oo & 31)] =
                            (_Float16)(acc[nt][r] + biasr[r]);
                    }
                }
            } else {                                  // v1: [c][tok]
#pragma unroll
                for (int nt = 0; nt < 4; ++nt) {
                    int n = nt * 16 + n15;
#pragma unroll
                    for (int r = 0; r < 8; ++r) {
                        int c = mt * 16 + r + 8 * hi - 512;
                        v1lds[c * NPAD + n] = (_Float16)(acc[nt][r] + biasr[r]);
                    }
                }
            }
        }
    }
    __syncthreads();

    // ---------------- Phase C: dwconv 3x3 + BN + GELU + residual -> v2T -----
    for (int i = tid; i < VDIM * NTOK; i += 256) {
        int c = i / NTOK, t = i % NTOK;
        int y = t / 7, xx = t % 7;
        float s = 0.0f;
#pragma unroll
        for (int ky = 0; ky < 3; ++ky)
#pragma unroll
            for (int kx = 0; kx < 3; ++kx) {
                int yy = y + ky - 1, xc = xx + kx - 1;
                if (yy >= 0 && yy < 7 && xc >= 0 && xc < 7)
                    s += dww[c * 9 + ky * 3 + kx] * (float)v1lds[c * NPAD + yy * 7 + xc];
            }
        float scale = bng[c] * rsqrtf(bnv[c] + 1e-5f);
        float d = (s - bnm[c]) * scale + bnb[c];
        float g = 0.5f * d * (1.0f + erff(d * 0.70710678118f));
        float v2 = (float)v1lds[c * NPAD + t] + g;
        v2T[(c >> 6) * 4096 + (c & 63) * NPAD + t] = (_Float16)v2;  // [h][dv][tok]
    }
    for (int i = tid; i < VDIM * (NPAD - NTOK); i += 256) {
        int c = i / (NPAD - NTOK), t = NTOK + i % (NPAD - NTOK);
        v2T[(c >> 6) * 4096 + (c & 63) * NPAD + t] = (_Float16)0.0f;
    }
    __syncthreads();

    // ---------------- Phase D: per-head QK^T + softmax (wave == head) -------
    {
        int h = wave;
        const _Float16* qh = qlds + h * 2048;   // [64][32]
        const _Float16* kh = klds + h * 2048;   // [64][32] == B^T layout
        v16h aq[4], bk[4];
#pragma unroll
        for (int t = 0; t < 4; ++t) {
            aq[t] = load_afrag(qh + t * 16 * KDIM, KDIM, lane);
            bk[t] = load_bfrag(kh + t * 16 * KDIM, KDIM, lane);
        }
        v8f s[4][4];
#pragma unroll
        for (int mt = 0; mt < 4; ++mt)
#pragma unroll
            for (int nt = 0; nt < 4; ++nt)
                s[mt][nt] = wmma16(aq[mt], bk[nt], v8f{});

        const float* bf = biasfull + h * 4096;  // [nk][mq] transposed
#pragma unroll
        for (int mt = 0; mt < 4; ++mt) {
#pragma unroll
            for (int nt = 0; nt < 4; ++nt) {
                const float* bp = bf + (nt * 16 + n15) * 64 + mt * 16 + 8 * hi;
#pragma unroll
                for (int r = 0; r < 8; ++r)
                    s[mt][nt][r] = s[mt][nt][r] * SCALEQ + bp[r];
            }
#pragma unroll
            for (int r = 0; r < 8; ++r) {
                float mx = fmaxf(fmaxf(s[mt][0][r], s[mt][1][r]),
                                 fmaxf(s[mt][2][r], s[mt][3][r]));
#pragma unroll
                for (int off = 1; off < 16; off <<= 1)
                    mx = fmaxf(mx, __shfl_xor(mx, off, 32));
                float e[4], sum = 0.0f;
#pragma unroll
                for (int nt = 0; nt < 4; ++nt) {
                    e[nt] = __expf(s[mt][nt][r] - mx);
                    sum += e[nt];
                }
#pragma unroll
                for (int off = 1; off < 16; off <<= 1)
                    sum += __shfl_xor(sum, off, 32);
                float inv = 1.0f / sum;
                int m = mt * 16 + r + 8 * hi;
#pragma unroll
                for (int nt = 0; nt < 4; ++nt)
                    attnP[h * 4096 + m * 64 + nt * 16 + n15] = (_Float16)(e[nt] * inv);
            }
        }
    }
    __syncthreads();

    // ---------------- Phase E: attn @ V  (wave == head) ---------------------
    {
        int h = wave;
        const _Float16* ph = attnP + h * 4096;  // A: [tok][tok]
        const _Float16* vh = v2T + h * 4096;    // B^T: [dv][tok]
#pragma unroll
        for (int mt = 0; mt < 4; ++mt) {
            v16h a0 = load_afrag(ph + mt * 16 * 64 + 0, 64, lane);
            v16h a1 = load_afrag(ph + mt * 16 * 64 + 32, 64, lane);
#pragma unroll
            for (int nt = 0; nt < 4; ++nt) {
                v8f acc = {};
                acc = wmma16(a0, load_bfrag(vh + nt * 16 * 64 + 0, 64, lane), acc);
                acc = wmma16(a1, load_bfrag(vh + nt * 16 * 64 + 32, 64, lane), acc);
#pragma unroll
                for (int r = 0; r < 8; ++r) {
                    int tok = mt * 16 + r + 8 * hi;
                    attnOut[tok * VDIM + h * 64 + nt * 16 + n15] = (_Float16)acc[r];
                }
            }
        }
    }
    __syncthreads();

    // ---------------- Phase F: proj  out = attnOut(49x512) @ projW^T --------
    // M = tokens (4 tiles), N = dim (24 tiles, 3 per wave, all simultaneous so
    // each A-fragment is loaded once) -> coalesced f32 stores.
    {
        float* outb = out + (size_t)blk * NTOK * DIMX;
        const int nt0 = wave, nt1 = wave + 8, nt2 = wave + 16;
        v8f acc[3][4] = {};
        for (int kk = 0; kk < 16; ++kk) {
            v16h b0 = load_bfrag(projf16 + nt0 * 16 * VDIM + kk * 32, VDIM, lane);
            v16h b1 = load_bfrag(projf16 + nt1 * 16 * VDIM + kk * 32, VDIM, lane);
            v16h b2 = load_bfrag(projf16 + nt2 * 16 * VDIM + kk * 32, VDIM, lane);
#pragma unroll
            for (int mt = 0; mt < 4; ++mt) {
                v16h a = load_afrag(attnOut + mt * 16 * VDIM + kk * 32, VDIM, lane);
                acc[0][mt] = wmma16(a, b0, acc[0][mt]);
                acc[1][mt] = wmma16(a, b1, acc[1][mt]);
                acc[2][mt] = wmma16(a, b2, acc[2][mt]);
            }
        }
#pragma unroll
        for (int j = 0; j < 3; ++j) {
            int nt = wave + 8 * j;
            float pbr = projb[nt * 16 + n15];       // per-lane bias (d = lane col)
#pragma unroll
            for (int mt = 0; mt < 4; ++mt)
#pragma unroll
                for (int r = 0; r < 8; ++r) {
                    int tok = mt * 16 + r + 8 * hi;
                    if (tok < NTOK)
                        outb[tok * DIMX + nt * 16 + n15] = acc[j][mt][r] + pbr;
                }
        }
    }
}

// ---------------------------------------------------------------------------
extern "C" void kernel_launch(void* const* d_in, const int* in_sizes, int n_in,
                              void* d_out, int out_size, void* d_ws, size_t ws_size,
                              hipStream_t stream) {
    const float* x           = (const float*)d_in[0];
    const float* pwconv_w    = (const float*)d_in[1];
    const float* pwconv_b    = (const float*)d_in[2];
    const float* dw_w        = (const float*)d_in[3];
    const float* bn_gamma    = (const float*)d_in[4];
    const float* bn_beta     = (const float*)d_in[5];
    const float* bn_mean     = (const float*)d_in[6];
    const float* bn_var      = (const float*)d_in[7];
    const float* attn_biases = (const float*)d_in[8];
    const float* proj_w      = (const float*)d_in[9];
    const float* proj_b      = (const float*)d_in[10];
    const int*   bias_idxs   = (const int*)d_in[11];

    char* ws = (char*)d_ws;
    _Float16* pwf16   = (_Float16*)(ws + WS_PW);
    _Float16* projf16 = (_Float16*)(ws + WS_PROJ);
    float*    biasful = (float*)(ws + WS_BIAS);

    prep_kernel<<<(1024 * DIMX + 255) / 256, 256, 0, stream>>>(
        pwconv_w, proj_w, attn_biases, bias_idxs, pwf16, projf16, biasful);

    const int B = in_sizes[0] / (NTOK * DIMX);   // 2048
    levit_attn_kernel<<<B, 256, 0, stream>>>(
        x, pwconv_b, dw_w, bn_gamma, bn_beta, bn_mean, bn_var,
        pwf16, projf16, biasful, proj_b, (float*)d_out);
}
